// FullSparseAttention_11776800325695
// MI455X (gfx1250) — matrix operand order repaired and checked by reading, not compiled, not run on hardware
//
#include <hip/hip_runtime.h>
#include <hip/hip_bf16.h>
#include <math.h>

typedef __bf16 bf16;
typedef __attribute__((ext_vector_type(16))) __bf16 v16bf;
typedef __attribute__((ext_vector_type(8)))  float  v8f;

#define D_MODEL 1024
#define NB      2048            // tokens per batch (L*K = 512*4)
#define BATCH   2
#define MTOT    (BATCH * NB)    // 4096 total token rows
#define NHEAD   16
#define HD      64
#define SCALE   0.125f          // HD^-0.5

// ---------------------------------------------------------------------------
// WMMA fragment helpers (wave32, V_WMMA_F32_16X16X32_BF16)
// A-matrix 16x32 bf16: lane L<16 holds row M=L, K = 0..7 (v0..3) and 16..23
// (v4..7); lanes 16..31 hold K = 8..15 and 24..31.  Both chunks contiguous.
// ---------------------------------------------------------------------------
__device__ __forceinline__ v16bf frag_a(const bf16* __restrict__ base, int ld, int lane) {
    int row = lane & 15;
    int kb  = (lane >> 4) * 8;
    const bf16* p = base + (size_t)row * ld + kb;
    v16bf a;
#pragma unroll
    for (int e = 0; e < 8; ++e) a[e] = p[e];
#pragma unroll
    for (int e = 0; e < 8; ++e) a[8 + e] = p[16 + e];
    return a;
}

// B-matrix 32x16 bf16: lane L<16 holds column N=L, K=0..7 and 16..23; lanes 16..31
// hold column N=L-16, K=8..15 and 24..31 (same K order as frag_a).
__device__ __forceinline__ v16bf frag_b(const bf16* __restrict__ base, int ld, int lane) {
    int col = lane & 15;
    int kb  = (lane >> 4) * 8;
    const bf16* p = base + (size_t)col * ld + kb;
    v16bf v;
#pragma unroll
    for (int e = 0; e < 8; ++e) v[e] = p[e];
#pragma unroll
    for (int e = 0; e < 8; ++e) v[8 + e] = p[16 + e];
    return v;
}

__device__ __forceinline__ v8f wmma_bf16(v16bf a, v16bf b, v8f c) {
    return __builtin_amdgcn_wmma_f32_16x16x32_bf16(
        /*neg_a=*/false, a, /*neg_b=*/false, b,
        /*c_mod=*/(short)0, c, /*reuse_a=*/false, /*reuse_b=*/false);
}

__device__ __forceinline__ float redmax16(float v) {
    v = fmaxf(v, __shfl_xor(v, 1));
    v = fmaxf(v, __shfl_xor(v, 2));
    v = fmaxf(v, __shfl_xor(v, 4));
    v = fmaxf(v, __shfl_xor(v, 8));
    return v;
}
__device__ __forceinline__ float redsum16(float v) {
    v += __shfl_xor(v, 1);
    v += __shfl_xor(v, 2);
    v += __shfl_xor(v, 4);
    v += __shfl_xor(v, 8);
    return v;
}

// ---------------------------------------------------------------------------
// 1) h + sinusoidal PE -> bf16
// ---------------------------------------------------------------------------
__global__ void k_prep_h(const float* __restrict__ h, bf16* __restrict__ hbf, int total) {
    int i = blockIdx.x * blockDim.x + threadIdx.x;
    if (i >= total) return;
    int d  = i & (D_MODEL - 1);
    int n  = (i >> 10) & (NB - 1);
    int i2 = d >> 1;
    float freq = __expf((-logf(10000.0f) / (float)D_MODEL) * (float)(2 * i2));
    float ang  = (float)n * freq;
    float pe   = (d & 1) ? __cosf(ang) : __sinf(ang);
    hbf[i] = (bf16)(h[i] + pe);
}

// 2) key bias = prob_bias_scale * log(probs + 1e-8)
__global__ void k_bias(const float* __restrict__ probs, const float* __restrict__ pbs,
                       float* __restrict__ biasLog, int total) {
    int i = blockIdx.x * blockDim.x + threadIdx.x;
    if (i < total) biasLog[i] = pbs[0] * __logf(probs[i] + 1e-8f);
}

// 3) f32 -> bf16 convert (weights)
__global__ void k_cvt(const float* __restrict__ src, bf16* __restrict__ dst, int total) {
    int i = blockIdx.x * blockDim.x + threadIdx.x;
    if (i < total) dst[i] = (bf16)src[i];
}

// ---------------------------------------------------------------------------
// 4) GEMM: C[M,Nc] = A[M,Kd] @ W[Nc,Kd]^T + bias
//    block 256 thr = 8 waves (2M x 4N), block tile 64x128, wave tile 32x32.
//    outMode 0: bf16 row-major; 1: bf16 transposed per-batch [b][col][n]
//    (for V -> Vt so attention's P*V B-fragment is key-contiguous);
//    2: f32 row-major (final output).
// ---------------------------------------------------------------------------
__global__ __launch_bounds__(256)
void k_gemm(const bf16* __restrict__ A, const bf16* __restrict__ W,
            const float* __restrict__ bias, void* __restrict__ Cout,
            int M, int Nc, int Kd, int outMode) {
    int lane = threadIdx.x & 31;
    int w    = threadIdx.x >> 5;
    int wm   = w >> 2, wn = w & 3;
    int m0   = blockIdx.y * 64 + wm * 32;
    int n0   = blockIdx.x * 128 + wn * 32;

    v8f acc[2][2] = {};
    for (int k0 = 0; k0 < Kd; k0 += 32) {
        v16bf a0 = frag_a(A + (size_t)m0 * Kd + k0, Kd, lane);
        v16bf a1 = frag_a(A + (size_t)(m0 + 16) * Kd + k0, Kd, lane);
        v16bf b0 = frag_b(W + (size_t)n0 * Kd + k0, Kd, lane);
        v16bf b1 = frag_b(W + (size_t)(n0 + 16) * Kd + k0, Kd, lane);
        acc[0][0] = wmma_bf16(a0, b0, acc[0][0]);
        acc[0][1] = wmma_bf16(a0, b1, acc[0][1]);
        acc[1][0] = wmma_bf16(a1, b0, acc[1][0]);
        acc[1][1] = wmma_bf16(a1, b1, acc[1][1]);
    }

    int col_l = lane & 15;
    int rbase = (lane >> 4) * 8;
#pragma unroll
    for (int i = 0; i < 2; ++i)
#pragma unroll
        for (int j = 0; j < 2; ++j)
#pragma unroll
            for (int r = 0; r < 8; ++r) {
                int row = m0 + i * 16 + rbase + r;
                int col = n0 + j * 16 + col_l;
                float v = acc[i][j][r] + (bias ? bias[col] : 0.0f);
                if (outMode == 0) {
                    ((bf16*)Cout)[(size_t)row * Nc + col] = (bf16)v;
                } else if (outMode == 1) {
                    int b = row >> 11, n = row & (NB - 1);
                    ((bf16*)Cout)[((size_t)b * Nc + col) * NB + n] = (bf16)v;
                } else {
                    ((float*)Cout)[(size_t)row * Nc + col] = v;
                }
            }
}

// ---------------------------------------------------------------------------
// 5) Flash attention: one wave per 16-row query tile, online softmax over all
//    2048 keys in chunks of 32.  4 WMMAs (QK^T) + 4 WMMAs (P*V) per chunk.
// ---------------------------------------------------------------------------
__global__ __launch_bounds__(128)
void k_attn(const bf16* __restrict__ Q, const bf16* __restrict__ Km,
            const bf16* __restrict__ Vt, const float* __restrict__ biasLog,
            bf16* __restrict__ O) {
    __shared__ bf16 ldsP[4][16][32];

    int lane  = threadIdx.x & 31;
    int w     = threadIdx.x >> 5;
    int qt    = blockIdx.x * 4 + w;
    int head  = blockIdx.y;
    int b     = blockIdx.z;
    int qrow0 = qt * 16;
    int col_l = lane & 15;
    int hi    = lane >> 4;

    const bf16* Qbase = Q + ((size_t)b * NB + qrow0) * D_MODEL + head * HD;
    v16bf aq0 = frag_a(Qbase + 0,  D_MODEL, lane);   // hd 0..31
    v16bf aq1 = frag_a(Qbase + 32, D_MODEL, lane);   // hd 32..63

    float m[8], l[8];
    v8f o[4] = {};
#pragma unroll
    for (int r = 0; r < 8; ++r) { m[r] = -1e30f; l[r] = 0.0f; }

    const float* biasB = biasLog + (size_t)b * NB;

    for (int kb = 0; kb < NB; kb += 32) {
        // ----- scores: S = Q K^T for 32 keys (two 16-key N-tiles) -----
        const bf16* Kb0 = Km + ((size_t)b * NB + kb) * D_MODEL + head * HD;
        const bf16* Kb1 = Kb0 + (size_t)16 * D_MODEL;
        v8f s0 = {}, s1 = {};
        s0 = wmma_bf16(aq0, frag_b(Kb0 + 0,  D_MODEL, lane), s0);
        s0 = wmma_bf16(aq1, frag_b(Kb0 + 32, D_MODEL, lane), s0);
        s1 = wmma_bf16(aq0, frag_b(Kb1 + 0,  D_MODEL, lane), s1);
        s1 = wmma_bf16(aq1, frag_b(Kb1 + 32, D_MODEL, lane), s1);

        float bias0 = biasB[kb + col_l];
        float bias1 = biasB[kb + 16 + col_l];

        // ----- online softmax across this 32-key chunk -----
        // C-layout: element r of this lane is row (hi*8 + r), col col_l.
#pragma unroll
        for (int r = 0; r < 8; ++r) {
            float x0   = s0[r] * SCALE + bias0;
            float x1   = s1[r] * SCALE + bias1;
            float mx   = redmax16(fmaxf(x0, x1));
            float mnew = fmaxf(m[r], mx);
            float alpha = __expf(m[r] - mnew);
            float p0    = __expf(x0 - mnew);
            float p1    = __expf(x1 - mnew);
            l[r] = l[r] * alpha + redsum16(p0 + p1);
            m[r] = mnew;
            o[0][r] *= alpha; o[1][r] *= alpha;
            o[2][r] *= alpha; o[3][r] *= alpha;
            // stage P row-major 16x32 in LDS for re-read in A-fragment order
            ldsP[w][hi * 8 + r][col_l]      = (bf16)p0;
            ldsP[w][hi * 8 + r][16 + col_l] = (bf16)p1;
        }
        asm volatile("s_wait_dscnt 0x0" ::: "memory");  // LDS RAW across lanes (lockstep wave)

        // reload P as bf16 A-fragment (16 rows x 32 keys)
        v16bf ap;
        {
            int row = col_l;
            int kq  = hi * 8;
#pragma unroll
            for (int e = 0; e < 8; ++e) ap[e] = ldsP[w][row][kq + e];
#pragma unroll
            for (int e = 0; e < 8; ++e) ap[8 + e] = ldsP[w][row][16 + kq + e];
        }

        // ----- O += P @ V  (Vt[b][d][n] is key-contiguous per hd column) -----
#pragma unroll
        for (int t = 0; t < 4; ++t) {
            const bf16* Vb = Vt + ((size_t)b * D_MODEL + head * HD + t * 16) * NB + kb;
            o[t] = wmma_bf16(ap, frag_b(Vb, NB, lane), o[t]);
        }
    }

    // ----- epilogue: normalize and store bf16 [B*N, D] -----
#pragma unroll
    for (int t = 0; t < 4; ++t)
#pragma unroll
        for (int r = 0; r < 8; ++r) {
            int row = qrow0 + hi * 8 + r;
            int col = head * HD + t * 16 + col_l;
            O[((size_t)b * NB + row) * D_MODEL + col] = (bf16)(o[t][r] / l[r]);
        }
}

// ---------------------------------------------------------------------------
// Host launcher
// ---------------------------------------------------------------------------
extern "C" void kernel_launch(void* const* d_in, const int* in_sizes, int n_in,
                              void* d_out, int out_size, void* d_ws, size_t ws_size,
                              hipStream_t stream) {
    const float* h     = (const float*)d_in[0];
    const float* probs = (const float*)d_in[1];
    const float* Wq    = (const float*)d_in[2];
    const float* bq    = (const float*)d_in[3];
    const float* Wk    = (const float*)d_in[4];
    const float* bk    = (const float*)d_in[5];
    const float* Wv    = (const float*)d_in[6];
    const float* bv    = (const float*)d_in[7];
    const float* Wo    = (const float*)d_in[8];
    const float* bo    = (const float*)d_in[9];
    const float* pbs   = (const float*)d_in[10];
    float* out         = (float*)d_out;

    // workspace carve-up (256B aligned)
    char* ws = (char*)d_ws;
    size_t off = 0;
    auto carve = [&](size_t bytes) { char* p = ws + off; off += (bytes + 255) & ~(size_t)255; return p; };
    const size_t actBytes = (size_t)MTOT * D_MODEL * sizeof(bf16);   // 8 MB
    const size_t wBytes   = (size_t)D_MODEL * D_MODEL * sizeof(bf16);// 2 MB
    bf16*  hbf   = (bf16*)carve(actBytes);
    bf16*  wqb   = (bf16*)carve(wBytes);
    bf16*  wkb   = (bf16*)carve(wBytes);
    bf16*  wvb   = (bf16*)carve(wBytes);
    bf16*  wob   = (bf16*)carve(wBytes);
    bf16*  Qb    = (bf16*)carve(actBytes);
    bf16*  Kb    = (bf16*)carve(actBytes);
    bf16*  Vtb   = (bf16*)carve(actBytes);
    bf16*  Ob    = (bf16*)carve(actBytes);
    float* biasL = (float*)carve((size_t)MTOT * sizeof(float));

    const int totalAct = MTOT * D_MODEL;     // 4194304
    const int totalW   = D_MODEL * D_MODEL;  // 1048576

    // 1) h + PE -> bf16 ; key bias ; weight converts
    k_prep_h<<<dim3((totalAct + 255) / 256), dim3(256), 0, stream>>>(h, hbf, totalAct);
    k_bias<<<dim3((MTOT + 255) / 256), dim3(256), 0, stream>>>(probs, pbs, biasL, MTOT);
    k_cvt<<<dim3((totalW + 255) / 256), dim3(256), 0, stream>>>(Wq, wqb, totalW);
    k_cvt<<<dim3((totalW + 255) / 256), dim3(256), 0, stream>>>(Wk, wkb, totalW);
    k_cvt<<<dim3((totalW + 255) / 256), dim3(256), 0, stream>>>(Wv, wvb, totalW);
    k_cvt<<<dim3((totalW + 255) / 256), dim3(256), 0, stream>>>(Wo, wob, totalW);

    // 2) projections: grid = (Nc/128, M/64)
    dim3 ggrid(D_MODEL / 128, MTOT / 64), gblk(256);
    k_gemm<<<ggrid, gblk, 0, stream>>>(hbf, wqb, bq, Qb,  MTOT, D_MODEL, D_MODEL, 0);
    k_gemm<<<ggrid, gblk, 0, stream>>>(hbf, wkb, bk, Kb,  MTOT, D_MODEL, D_MODEL, 0);
    k_gemm<<<ggrid, gblk, 0, stream>>>(hbf, wvb, bv, Vtb, MTOT, D_MODEL, D_MODEL, 1); // transposed

    // 3) attention: 128 q-tiles / 4 waves per block
    k_attn<<<dim3(NB / 16 / 4, NHEAD, BATCH), dim3(128), 0, stream>>>(Qb, Kb, Vtb, biasL, Ob);

    // 4) output projection -> f32 d_out
    k_gemm<<<ggrid, gblk, 0, stream>>>(Ob, wob, bo, out, MTOT, D_MODEL, D_MODEL, 2);

    (void)in_sizes; (void)n_in; (void)out_size; (void)ws_size;
}
